// MER_49864570306679
// MI455X (gfx1250) — compile-verified, matrix operand order checked by reading, not run on hardware
//
#include <hip/hip_runtime.h>
#include <hip/hip_bf16.h>

typedef __attribute__((ext_vector_type(16))) _Float16 v16h;
typedef __attribute__((ext_vector_type(8)))  float    v8f;

#define B_   64
#define T_   512
#define H_   256
#define A_   100
#define L_   10
#define EW_  300
#define ED_  64
#define KP_  320      // EW padded to multiple of 32
#define NB_  1136     // 1024 (gates) + 112 (word_proj padded)
#define AP_  112
#define NWG  8

#define WMMA_F16(a, b, c) \
  __builtin_amdgcn_wmma_f32_16x16x32_f16(false, (a), false, (b), (short)0, (c), false, false)

__device__ __forceinline__ float sigf(float x) { return 1.0f / (1.0f + __expf(-x)); }

// Load a 16x32 f16 WMMA fragment (A from row-major [m][k], or B from [n][k] "B^T" storage).
// gfx1250 layout: lane L holds row (L&15); element e maps to k = 16*(e>>3) + 8*(L>>4) + (e&7).
__device__ __forceinline__ v16h ld_frag(const _Float16* __restrict__ base, int ld, int r0, int k0) {
  const int lane = threadIdx.x & 31;
  const _Float16* p = base + (size_t)(r0 + (lane & 15)) * ld + k0 + ((lane >> 4) << 3);
  v16h v;
  ((uint4*)&v)[0] = *(const uint4*)(p);
  ((uint4*)&v)[1] = *(const uint4*)(p + 16);
  return v;
}

// ---------------- Kernel 1: gather word embeddings -> f16, K padded to 320 ----------------
__global__ void k_embed(const int* __restrict__ text, const float* __restrict__ w2v,
                        _Float16* __restrict__ E) {
  const int row = blockIdx.x;                       // b*T + t
  const int idx = text[row];
  const float* src = w2v + (size_t)idx * EW_;
  for (int k = threadIdx.x; k < KP_; k += blockDim.x) {
    float v = (k < EW_) ? src[k] : 0.0f;
    E[(size_t)row * KP_ + k] = (_Float16)v;
  }
}

// ---------------- Kernel 2: pack weights into f16 B-layout ([n][k]) buffers ----------------
__global__ void k_pack(const float* __restrict__ W_ac, const float* __restrict__ W_ih,
                       const float* __restrict__ W_hh,
                       _Float16* __restrict__ Wbig, _Float16* __restrict__ Whh,
                       _Float16* __restrict__ Wihu, _Float16* __restrict__ WhT) {
  int id = blockIdx.x * blockDim.x + threadIdx.x;
  const int S0 = NB_ * KP_;      // 363520: fused [W_ih word part | W_w^T], K=320
  const int S1 = 1024 * 256;     // 262144: W_hh (already [n][k])
  const int S2 = 1024 * 64;      // 65536 : W_ih u-part
  const int S3 = AP_ * 256;      // 28672 : W_h^T padded to 112 cols
  if (id < S0) {
    int n = id / KP_, k = id % KP_;
    float v = 0.0f;
    if (n < 1024) { if (k < EW_) v = W_ih[(size_t)n * 364 + k]; }
    else { int a = n - 1024; if (a < A_ && k < EW_) v = W_ac[(size_t)(ED_ + k) * A_ + a]; }
    Wbig[id] = (_Float16)v;
    return;
  }
  id -= S0;
  if (id < S1) { Whh[id] = (_Float16)W_hh[id]; return; }
  id -= S1;
  if (id < S2) {
    int n = id / 64, k = id % 64;
    Wihu[id] = (_Float16)W_ih[(size_t)n * 364 + EW_ + k];
    return;
  }
  id -= S2;
  if (id < S3) {
    int n = id / 256, k = id % 256;
    float v = (n < A_) ? W_ac[(size_t)(ED_ + EW_ + k) * A_ + n] : 0.0f;
    WhT[id] = (_Float16)v;
  }
}

// ---------------- Kernel 3: dom_emb gather + dom_proj = dom_emb @ W_d ----------------
__global__ void k_dom(const int* __restrict__ domains, const float* __restrict__ dtab,
                      const float* __restrict__ W_ac,
                      float* __restrict__ domE, float* __restrict__ domP) {
  const int bl = blockIdx.x;   // b*L + l
  __shared__ float e[ED_];
  const int d = domains[bl];
  if (threadIdx.x < ED_) {
    float v = dtab[(size_t)d * ED_ + threadIdx.x];
    e[threadIdx.x] = v;
    domE[(size_t)bl * ED_ + threadIdx.x] = v;
  }
  __syncthreads();
  if (threadIdx.x < A_) {
    float acc = 0.0f;
    for (int k = 0; k < ED_; ++k) acc = fmaf(e[k], W_ac[(size_t)k * A_ + threadIdx.x], acc);
    domP[(size_t)bl * A_ + threadIdx.x] = acc;
  }
}

// ---------------- Kernel 4: big precompute GEMM  Gw = E @ Wbig^T (+bias on gate cols) ------
// M=32768, N=1136, K=320. One 16x16 tile per wave, 10 WMMA each.
__global__ void __launch_bounds__(256) k_gemm_pre(
    const _Float16* __restrict__ E, const _Float16* __restrict__ Wbig,
    const float* __restrict__ b_ih, const float* __restrict__ b_hh,
    float* __restrict__ Gw) {
  const int NT = NB_ / 16;  // 71
  const int tile = blockIdx.x * 8 + (threadIdx.x >> 5);
  const int mt = tile / NT, nt = tile % NT;
  const int m0 = mt * 16, n0 = nt * 16;
  v8f acc = {0.f, 0.f, 0.f, 0.f, 0.f, 0.f, 0.f, 0.f};
  for (int ks = 0; ks < KP_ / 32; ++ks) {
    v16h a = ld_frag(E, KP_, m0, ks * 32);
    v16h b = ld_frag(Wbig, KP_, n0, ks * 32);
    acc = WMMA_F16(a, b, acc);
  }
  const int lane = threadIdx.x & 31;
  const int half = lane >> 4, n = n0 + (lane & 15);
  const float bias = (n < 1024) ? (b_ih[n] + b_hh[n]) : 0.0f;
#pragma unroll
  for (int r = 0; r < 8; ++r) {
    int m = m0 + r + 8 * half;
    Gw[(size_t)m * NB_ + n] = acc[r] + bias;
  }
}

// ---------------- Kernel 5: persistent sequential LSTM+attention over T steps -------------
// 8 WGs x 256 threads. WG wg owns H columns [wg*32, wg*32+32). Each wave owns one
// (b-tile mt, 16-col group hs) pair and keeps its c-state in registers (WMMA C layout).
__global__ void __launch_bounds__(256, 1) k_seq(
    const float* __restrict__ Gw, const _Float16* __restrict__ Whh,
    const _Float16* __restrict__ Wihu, const _Float16* __restrict__ WhT,
    const float* __restrict__ domE, const float* __restrict__ domP,
    const int* __restrict__ dmask, const float* __restrict__ Vac,
    _Float16* __restrict__ h16,          // [2][B][H] double buffer (pre-zeroed)
    float* __restrict__ outHid,          // [B][T][H]
    unsigned* __restrict__ bar) {
  __shared__ float attS[B_ * AP_];
  __shared__ __align__(16) _Float16 uS[B_ * ED_];
  __shared__ float scoreS[B_ * 12];

  const int wg   = blockIdx.x;
  const int tid  = threadIdx.x;
  const int wave = tid >> 5;
  const int lane = tid & 31;
  const int half = lane >> 4;
  const int nl   = lane & 15;
  const int mt   = wave >> 1;              // b tile (0..3)
  const int hbase = wg * 32 + (wave & 1) * 16;
  const v8f vzero = {0.f, 0.f, 0.f, 0.f, 0.f, 0.f, 0.f, 0.f};

  v8f creg = vzero;                         // cell state, WMMA C layout

  for (int t = 0; t < T_; ++t) {
    const _Float16* hcur = h16 + (size_t)(t & 1) * (B_ * H_);
    _Float16* hnext      = h16 + (size_t)((t + 1) & 1) * (B_ * H_);

    // ---- Phase A: att = h @ W_h + word_proj[t]   (4 x 7 tiles, K=256) ----
    for (int ti = wave; ti < 28; ti += 8) {
      const int mt2 = ti / 7, nt2 = ti % 7;
      v8f acc = vzero;
      for (int ks = 0; ks < H_ / 32; ++ks) {
        v16h a = ld_frag(hcur, H_, mt2 * 16, ks * 32);
        v16h b = ld_frag(WhT, H_, nt2 * 16, ks * 32);
        acc = WMMA_F16(a, b, acc);
      }
      const int n = nt2 * 16 + nl;
#pragma unroll
      for (int r = 0; r < 8; ++r) {
        int m = mt2 * 16 + r + 8 * half;
        attS[m * AP_ + n] = acc[r] + Gw[((size_t)m * T_ + t) * NB_ + 1024 + n];
      }
    }
    __syncthreads();

    // ---- Phase B: scores -> softmax(L) -> u = alpha @ dom_emb ----
    for (int p = tid; p < B_ * L_; p += 256) {
      const int b = p / L_;
      float s;
      if (dmask[p] != 0) s = -__builtin_inff();
      else {
        s = 0.0f;
        const float* dp = domP + (size_t)p * A_;
        const float* at = attS + b * AP_;
        for (int a = 0; a < A_; ++a) s = fmaf(tanhf(dp[a] + at[a]), Vac[a], s);
      }
      scoreS[b * 12 + (p % L_)] = s;
    }
    __syncthreads();
    if (tid < B_) {
      float mx = -__builtin_inff();
      for (int l = 0; l < L_; ++l) mx = fmaxf(mx, scoreS[tid * 12 + l]);
      float ex[L_], sum = 0.0f;
      for (int l = 0; l < L_; ++l) { ex[l] = __expf(scoreS[tid * 12 + l] - mx); sum += ex[l]; }
      const float inv = 1.0f / sum;
      for (int l = 0; l < L_; ++l) scoreS[tid * 12 + l] = ex[l] * inv;
    }
    __syncthreads();
    for (int i = tid; i < B_ * ED_; i += 256) {
      const int b = i >> 6, d = i & 63;
      float acc = 0.0f;
      for (int l = 0; l < L_; ++l)
        acc = fmaf(scoreS[b * 12 + l], domE[((size_t)b * L_ + l) * ED_ + d], acc);
      uS[i] = (_Float16)acc;
    }
    __syncthreads();

    // ---- Phase C: gates = Gw[t] + h@Whh^T + u@Wihu^T, then in-lane LSTM pointwise ----
    v8f accG[4];
#pragma unroll
    for (int g = 0; g < 4; ++g) accG[g] = vzero;
    for (int ks = 0; ks < H_ / 32; ++ks) {
      v16h a = ld_frag(hcur, H_, mt * 16, ks * 32);
#pragma unroll
      for (int g = 0; g < 4; ++g) {
        v16h b = ld_frag(Whh, H_, g * 256 + hbase, ks * 32);
        accG[g] = WMMA_F16(a, b, accG[g]);
      }
    }
#pragma unroll
    for (int ks = 0; ks < ED_ / 32; ++ks) {
      v16h a = ld_frag(uS, ED_, mt * 16, ks * 32);
#pragma unroll
      for (int g = 0; g < 4; ++g) {
        v16h b = ld_frag(Wihu, ED_, g * 256 + hbase, ks * 32);
        accG[g] = WMMA_F16(a, b, accG[g]);
      }
    }
    {
      const int hidx = hbase + nl;
#pragma unroll
      for (int r = 0; r < 8; ++r) {
        const int m = mt * 16 + r + 8 * half;        // batch index
        const size_t gb = ((size_t)m * T_ + t) * NB_;
        const float xi = accG[0][r] + Gw[gb + 0 * 256 + hidx];
        const float xf = accG[1][r] + Gw[gb + 1 * 256 + hidx];
        const float xg = accG[2][r] + Gw[gb + 2 * 256 + hidx];
        const float xo = accG[3][r] + Gw[gb + 3 * 256 + hidx];
        const float c  = creg[r];
        const float cn = sigf(xf) * c + sigf(xi) * tanhf(xg);
        const float hn = sigf(xo) * tanhf(cn);
        creg[r] = cn;
        outHid[((size_t)m * T_ + t) * H_ + hidx] = hn;
        hnext[m * H_ + hidx] = (_Float16)hn;
      }
    }

    // ---- grid-wide barrier (8 resident WGs) ----
    __threadfence();
    __syncthreads();
    if (tid == 0) {
      __hip_atomic_fetch_add(bar, 1u, __ATOMIC_RELEASE, __HIP_MEMORY_SCOPE_AGENT);
      const unsigned target = (unsigned)NWG * (unsigned)(t + 1);
      while (__hip_atomic_load(bar, __ATOMIC_ACQUIRE, __HIP_MEMORY_SCOPE_AGENT) < target)
        __builtin_amdgcn_s_sleep(2);
    }
    __syncthreads();
  }
}

// ---------------- Kernel 6: last_hidden = one-hot(mask) @ hidden_states ----------------
__global__ void k_last(const float* __restrict__ mask, const float* __restrict__ hid,
                       float* __restrict__ out) {
  const int b = blockIdx.x, h = threadIdx.x;
  float acc = 0.0f;
  for (int t = 0; t < T_; ++t)
    acc = fmaf(mask[b * T_ + t], hid[((size_t)b * T_ + t) * H_ + h], acc);
  out[b * H_ + h] = acc;
}

extern "C" void kernel_launch(void* const* d_in, const int* in_sizes, int n_in,
                              void* d_out, int out_size, void* d_ws, size_t ws_size,
                              hipStream_t stream) {
  (void)in_sizes; (void)n_in; (void)out_size; (void)ws_size;
  const int*   text    = (const int*)d_in[0];
  const int*   domains = (const int*)d_in[1];
  const int*   dmask   = (const int*)d_in[2];
  const float* slmask  = (const float*)d_in[3];
  const float* w2v     = (const float*)d_in[4];
  const float* dtab    = (const float*)d_in[5];
  const float* W_ac    = (const float*)d_in[6];
  const float* V_ac    = (const float*)d_in[7];
  const float* W_ih    = (const float*)d_in[8];
  const float* W_hh    = (const float*)d_in[9];
  const float* b_ih    = (const float*)d_in[10];
  const float* b_hh    = (const float*)d_in[11];

  char* ws = (char*)d_ws;
  size_t off = 0;
  auto alloc = [&](size_t bytes) { size_t o = off; off = (off + bytes + 255) & ~(size_t)255; return o; };

  const size_t off_E    = alloc((size_t)B_ * T_ * KP_ * 2);     // f16 embeddings
  const size_t off_Wbig = alloc((size_t)NB_ * KP_ * 2);
  const size_t off_Whh  = alloc((size_t)1024 * 256 * 2);
  const size_t off_Wihu = alloc((size_t)1024 * 64 * 2);
  const size_t off_WhT  = alloc((size_t)AP_ * 256 * 2);
  const size_t off_domE = alloc((size_t)B_ * L_ * ED_ * 4);
  const size_t off_domP = alloc((size_t)B_ * L_ * A_ * 4);
  const size_t off_h16  = alloc((size_t)2 * B_ * H_ * 2);
  const size_t off_bar  = alloc(256);
  const size_t off_Gw   = alloc((size_t)B_ * T_ * NB_ * 4);     // ~149 MB

  _Float16* E    = (_Float16*)(ws + off_E);
  _Float16* Wbig = (_Float16*)(ws + off_Wbig);
  _Float16* Whh  = (_Float16*)(ws + off_Whh);
  _Float16* Wihu = (_Float16*)(ws + off_Wihu);
  _Float16* WhT  = (_Float16*)(ws + off_WhT);
  float*    domE = (float*)(ws + off_domE);
  float*    domP = (float*)(ws + off_domP);
  _Float16* h16  = (_Float16*)(ws + off_h16);
  unsigned* bar  = (unsigned*)(ws + off_bar);
  float*    Gw   = (float*)(ws + off_Gw);

  float* outHid  = (float*)d_out;
  float* outLast = outHid + (size_t)B_ * T_ * H_;

  // zero h0 (both double-buffer halves) and the grid-barrier counter each call
  hipMemsetAsync(h16, 0, (size_t)2 * B_ * H_ * 2, stream);
  hipMemsetAsync(bar, 0, 256, stream);

  k_embed<<<B_ * T_, 128, 0, stream>>>(text, w2v, E);
  {
    const int total = NB_ * KP_ + 1024 * 256 + 1024 * 64 + AP_ * 256;
    k_pack<<<(total + 255) / 256, 256, 0, stream>>>(W_ac, W_ih, W_hh, Wbig, Whh, Wihu, WhT);
  }
  k_dom<<<B_ * L_, 128, 0, stream>>>(domains, dtab, W_ac, domE, domP);
  {
    const int tiles = (B_ * T_ / 16) * (NB_ / 16);   // 2048 * 71 = 145408, /8 waves
    k_gemm_pre<<<tiles / 8, 256, 0, stream>>>(E, Wbig, b_ih, b_hh, Gw);
  }
  k_seq<<<NWG, 256, 0, stream>>>(Gw, Whh, Wihu, WhT, domE, domP, dmask, V_ac, h16, outHid, bar);
  k_last<<<B_, H_, 0, stream>>>(slmask, outHid, outLast);
}
